// TransformerBlock_35553739276551
// MI455X (gfx1250) — matrix operand order repaired and checked by reading, not compiled, not run on hardware
//
#include <hip/hip_runtime.h>
#include <hip/hip_bf16.h>
#include <math.h>
#include <stdint.h>

#define DMODEL 1024
#define DFF    4096
#define NHEADS 16
#define DHEAD  64
#define SEQB   2
#define SEQT   2048
#define MROWS  (SEQB*SEQT)     /* 4096 */
#define QKV_LD (3*DMODEL)      /* 3072 */

typedef float  v8f   __attribute__((ext_vector_type(8)));
typedef float  f32x4 __attribute__((ext_vector_type(4)));
typedef __bf16 v16bf __attribute__((ext_vector_type(16)));

union Frag16 { v16bf v; f32x4 q[2]; };

static __device__ inline f32x4 ld16(const __bf16* p) {
    return *reinterpret_cast<const f32x4*>(p);
}

// async global->LDS 16B copy (per lane), tracked by ASYNCcnt.
// VDST carries the wave-relative LDS byte address (low 32 bits of the generic
// address of a __shared__ object, per the flat-aperture truncation rule).
static __device__ inline void async_ld16(const __bf16* gp, __bf16* lp) {
    unsigned lds = (unsigned)(uintptr_t)lp;
    unsigned long long ga = (unsigned long long)(uintptr_t)gp;
    asm volatile("global_load_async_to_lds_b128 %0, %1, off"
                 :: "v"(lds), "v"(ga) : "memory");
}

// ---------------------------------------------------------------- cast f32->bf16
__global__ void cast_f32_bf16(const float* __restrict__ in, __bf16* __restrict__ out, size_t n) {
    size_t i = (size_t)blockIdx.x * blockDim.x + threadIdx.x;
    size_t stride = (size_t)gridDim.x * blockDim.x;
    for (; i < n; i += stride) out[i] = (__bf16)in[i];
}

// ------------------------------------------- transpose+cast: W[K][N] f32 -> Wt[N][K] bf16
__global__ __launch_bounds__(256)
void transpose_cast(const float* __restrict__ W, __bf16* __restrict__ Wt, int K, int N) {
    __shared__ float tile[32][33];
    const int kb = blockIdx.y * 32, nb = blockIdx.x * 32;
    const int tx = threadIdx.x, ty = threadIdx.y;   // 32 x 8
    #pragma unroll
    for (int r = 0; r < 32; r += 8)
        tile[ty + r][tx] = W[(size_t)(kb + ty + r) * N + nb + tx];
    __syncthreads();
    #pragma unroll
    for (int r = 0; r < 32; r += 8)
        Wt[(size_t)(nb + ty + r) * K + kb + tx] = (__bf16)tile[tx][ty + r];
}

// ---------------------------------------------------------------- WMMA GEMM
// C[M][N] = A[M][K](bf16) * Bt[N][K](bf16)^T + bias, epilogue per EPI:
//   EPI 0: bias, write bf16      EPI 1: bias+GELU, write bf16      EPI 2: bias, write f32
// Double-buffered LDS staging via async global->LDS loads (ASYNCcnt).
template<int EPI>
__global__ __launch_bounds__(256)
void gemm_bf16_wmma(const __bf16* __restrict__ A, const __bf16* __restrict__ Bt,
                    const float* __restrict__ bias,
                    float* __restrict__ Cf, __bf16* __restrict__ Cb,
                    int Mdim, int Ndim, int Kdim)
{
    constexpr int BM = 128, BN = 64, BK = 64, LDA = BK + 16, LDB = BK + 16;
    __shared__ __bf16 sA[2][BM * LDA];
    __shared__ __bf16 sB[2][BN * LDB];

    const int tid  = threadIdx.x;
    const int lane = tid & 31, wave = tid >> 5;
    const int hlf  = lane >> 4, lmod = lane & 15;
    const int wm   = wave >> 1, wn = wave & 1;
    const int m0   = blockIdx.y * BM, n0 = blockIdx.x * BN;

    v8f acc[2][2];
    #pragma unroll
    for (int a = 0; a < 2; a++)
        #pragma unroll
        for (int b = 0; b < 2; b++)
            #pragma unroll
            for (int e = 0; e < 8; e++) acc[a][b][e] = 0.f;

    // stage one K-tile (6 async 16B ops per thread)
    auto stage_async = [&](int buf, int k0) {
        #pragma unroll
        for (int i = 0; i < 4; i++) {
            int c = tid + i * 256;
            int r = c >> 3, ch = c & 7;
            async_ld16(A + (size_t)(m0 + r) * Kdim + k0 + ch * 8, &sA[buf][r * LDA + ch * 8]);
        }
        #pragma unroll
        for (int i = 0; i < 2; i++) {
            int c = tid + i * 256;
            int r = c >> 3, ch = c & 7;
            async_ld16(Bt + (size_t)(n0 + r) * Kdim + k0 + ch * 8, &sB[buf][r * LDB + ch * 8]);
        }
    };

    const int nk = Kdim / BK;
    stage_async(0, 0);

    for (int kt = 0; kt < nk; kt++) {
        const int cur = kt & 1;
        if (kt + 1 < nk) {
            stage_async(cur ^ 1, (kt + 1) * BK);
            // 6 newly issued outstanding allowed; current tile's 6 have landed
            asm volatile("s_wait_asynccnt 0x6" ::: "memory");
        } else {
            asm volatile("s_wait_asynccnt 0x0" ::: "memory");
        }
        __syncthreads();

        // load ALL fragments for both K-steps first, then burst the WMMAs
        Frag16 af[2][2], bfr[2][2];
        #pragma unroll
        for (int kk = 0; kk < 2; kk++) {
            #pragma unroll
            for (int im = 0; im < 2; im++) {
                int row = wm * 32 + im * 16 + lmod;
                int col = kk * 32 + hlf * 8;          // A: slots0-7 = K+half*8, slots8-15 = +16
                af[kk][im].q[0] = ld16(&sA[cur][row * LDA + col]);
                af[kk][im].q[1] = ld16(&sA[cur][row * LDA + col + 16]);
            }
            #pragma unroll
            for (int in = 0; in < 2; in++) {
                int row = wn * 32 + in * 16 + lmod;
                int col = kk * 32 + hlf * 8;          // B: slots0-7 = K+half*8, slots8-15 = +16
                bfr[kk][in].q[0] = ld16(&sB[cur][row * LDB + col]);
                bfr[kk][in].q[1] = ld16(&sB[cur][row * LDB + col + 16]);
            }
        }
        #pragma unroll
        for (int kk = 0; kk < 2; kk++)
            #pragma unroll
            for (int im = 0; im < 2; im++)
                #pragma unroll
                for (int in = 0; in < 2; in++)
                    acc[im][in] = __builtin_amdgcn_wmma_f32_16x16x32_bf16(
                        false, af[kk][im].v, false, bfr[kk][in].v, (short)0, acc[im][in], false, false);

        __syncthreads();   // all waves done with buffer `cur` before it is re-filled
    }

    // epilogue: C layout M = j + 8*half, N = lane%16
    #pragma unroll
    for (int im = 0; im < 2; im++)
        #pragma unroll
        for (int in = 0; in < 2; in++) {
            int gn = n0 + wn * 32 + in * 16 + lmod;
            float bv = bias[gn];
            #pragma unroll
            for (int j = 0; j < 8; j++) {
                int gm = m0 + wm * 32 + im * 16 + j + 8 * hlf;
                float v = acc[im][in][j] + bv;
                if (EPI == 1) v = 0.5f * v * (1.0f + erff(v * 0.70710678118f));
                if (EPI == 2) Cf[(size_t)gm * Ndim + gn] = v;
                else          Cb[(size_t)gm * Ndim + gn] = (__bf16)v;
            }
        }
}

// ---------------------------------------------------------------- flash attention
__global__ __launch_bounds__(256)
void attention_wmma(const __bf16* __restrict__ qkv, __bf16* __restrict__ out)
{
    constexpr int LDK = DHEAD + 8;            // 72 elems, 144B rows (16B aligned)
    __shared__ __bf16 sK[64 * LDK];           // K tile [kv][d]  (== B^T layout for Q*K^T)
    __shared__ __bf16 sV[DHEAD * LDK];        // V^T tile [d][kv] (== B^T layout for P*V)
    __shared__ __bf16 sP[8 * 16 * LDK];       // per-wave P tile [q][kv]

    const int tid  = threadIdx.x;
    const int lane = tid & 31, wave = tid >> 5;
    const int hlf  = lane >> 4, lmod = lane & 15;
    const int bh   = blockIdx.x;
    const int b    = bh >> 4, h = bh & 15;
    const int qbase0 = blockIdx.y * 128;
    const int qbase  = qbase0 + wave * 16;

    const size_t rowbase = (size_t)b * SEQT;
    const int qcol = h * DHEAD;

    // Q fragments (2 WMMA K-steps of 32) held in registers for the whole kernel
    Frag16 qf[2];
    #pragma unroll
    for (int kk = 0; kk < 2; kk++) {
        int row = qbase + lmod;
        const __bf16* p = qkv + (rowbase + row) * QKV_LD + qcol + kk * 32 + hlf * 8;
        qf[kk].q[0] = ld16(p);
        qf[kk].q[1] = ld16(p + 16);
    }

    v8f  o[4];
    float m[8], l[8];
    #pragma unroll
    for (int t = 0; t < 4; t++)
        #pragma unroll
        for (int e = 0; e < 8; e++) o[t][e] = 0.f;
    #pragma unroll
    for (int j = 0; j < 8; j++) { m[j] = -3.0e38f; l[j] = 0.f; }

    const int nkt = (qbase0 >> 6) + 2;        // causal: k tiles with k0 <= qbase0+127
    __bf16* myP = &sP[wave * 16 * LDK];

    for (int kt = 0; kt < nkt; kt++) {
        const int k0 = kt * 64;
        // stage K tile [kv][d] asynchronously (overlaps with V transpose below)
        #pragma unroll
        for (int i = 0; i < 2; i++) {
            int c = tid + i * 256;
            int r = c >> 3, ch = c & 7;
            async_ld16(qkv + (rowbase + k0 + r) * QKV_LD + DMODEL + qcol + ch * 8,
                       &sK[r * LDK + ch * 8]);
        }
        // stage V tile transposed -> sV[d][kv]
        #pragma unroll
        for (int i = 0; i < 2; i++) {
            int c = tid + i * 256;
            int r = c >> 3, ch = c & 7;    // r = kv row, ch*8 = d chunk
            f32x4 v = ld16(qkv + (rowbase + k0 + r) * QKV_LD + 2 * DMODEL + qcol + ch * 8);
            const __bf16* e = reinterpret_cast<const __bf16*>(&v);
            #pragma unroll
            for (int d = 0; d < 8; d++) sV[(ch * 8 + d) * LDK + r] = e[d];
        }
        asm volatile("s_wait_asynccnt 0x0" ::: "memory");
        __syncthreads();

        // wave-uniform causal skip; force a scalar branch so EXEC stays all-1s
        if (__builtin_amdgcn_readfirstlane((int)(k0 <= qbase + 15))) {
            // S = Q*K^T over 4 kv n-tiles: per K-step load 4 B-fragments, then burst WMMAs
            v8f s[4];
            #pragma unroll
            for (int t = 0; t < 4; t++)
                #pragma unroll
                for (int e = 0; e < 8; e++) s[t][e] = 0.f;
            #pragma unroll
            for (int kk = 0; kk < 2; kk++) {
                Frag16 bfr[4];
                #pragma unroll
                for (int nt = 0; nt < 4; nt++) {
                    const __bf16* p = &sK[(nt * 16 + lmod) * LDK + kk * 32 + hlf * 8];
                    bfr[nt].q[0] = ld16(p); bfr[nt].q[1] = ld16(p + 16);
                }
                #pragma unroll
                for (int nt = 0; nt < 4; nt++)
                    s[nt] = __builtin_amdgcn_wmma_f32_16x16x32_bf16(
                        false, qf[kk].v, false, bfr[nt].v, (short)0, s[nt], false, false);
            }
            // scale + causal mask + online softmax
            float ps[4][8];
            #pragma unroll
            for (int nt = 0; nt < 4; nt++) {
                int gk = k0 + nt * 16 + lmod;
                #pragma unroll
                for (int j = 0; j < 8; j++) {
                    int gq = qbase + j + 8 * hlf;
                    float sc = s[nt][j] * 0.125f;   // 1/sqrt(64)
                    ps[nt][j] = (gk <= gq) ? sc : -3.0e38f;
                }
            }
            #pragma unroll
            for (int j = 0; j < 8; j++) {
                float mx = fmaxf(fmaxf(ps[0][j], ps[1][j]), fmaxf(ps[2][j], ps[3][j]));
                #pragma unroll
                for (int off = 1; off < 16; off <<= 1) mx = fmaxf(mx, __shfl_xor(mx, off, 32));
                float mn = fmaxf(m[j], mx);
                float alpha = __expf(m[j] - mn);
                m[j] = mn;
                float psum = 0.f;
                #pragma unroll
                for (int nt = 0; nt < 4; nt++) {
                    float p = __expf(ps[nt][j] - mn);
                    ps[nt][j] = p; psum += p;
                }
                #pragma unroll
                for (int off = 1; off < 16; off <<= 1) psum += __shfl_xor(psum, off, 32);
                l[j] = l[j] * alpha + psum;
                #pragma unroll
                for (int t = 0; t < 4; t++) o[t][j] *= alpha;
            }
            // P -> LDS (bf16); same-wave DS ops are in-order
            #pragma unroll
            for (int nt = 0; nt < 4; nt++)
                #pragma unroll
                for (int j = 0; j < 8; j++)
                    myP[(j + 8 * hlf) * LDK + nt * 16 + lmod] = (__bf16)ps[nt][j];
            // O += P @ V: batch fragment loads per K-step, then burst WMMAs
            #pragma unroll
            for (int kk = 0; kk < 2; kk++) {
                Frag16 af, bfr[4];
                const __bf16* pa = &myP[lmod * LDK + kk * 32 + hlf * 8];
                af.q[0] = ld16(pa); af.q[1] = ld16(pa + 16);
                #pragma unroll
                for (int nt2 = 0; nt2 < 4; nt2++) {
                    const __bf16* pb = &sV[(nt2 * 16 + lmod) * LDK + kk * 32 + hlf * 8];
                    bfr[nt2].q[0] = ld16(pb); bfr[nt2].q[1] = ld16(pb + 16);
                }
                #pragma unroll
                for (int nt2 = 0; nt2 < 4; nt2++)
                    o[nt2] = __builtin_amdgcn_wmma_f32_16x16x32_bf16(
                        false, af.v, false, bfr[nt2].v, (short)0, o[nt2], false, false);
            }
        }
        __syncthreads();
    }

    // normalize and write attention output (bf16, [B,T,H*64] layout)
    #pragma unroll
    for (int nt2 = 0; nt2 < 4; nt2++)
        #pragma unroll
        for (int j = 0; j < 8; j++) {
            int gq = qbase + j + 8 * hlf;
            float v = o[nt2][j] / l[j];
            out[(rowbase + gq) * DMODEL + qcol + nt2 * 16 + lmod] = (__bf16)v;
        }
}

// ---------------------------------------------------------------- residual + LayerNorm
__global__ __launch_bounds__(256)
void add_ln(const float* __restrict__ X, const float* __restrict__ Y,
            const float* __restrict__ g, const float* __restrict__ bta,
            float* __restrict__ outF, __bf16* __restrict__ outB)
{
    const int row = blockIdx.x;
    const int tid = threadIdx.x;
    __shared__ float red[2][8];
    float v[4];
    float s = 0.f, ss = 0.f;
    #pragma unroll
    for (int c = 0; c < 4; c++) {
        int idx = tid + c * 256;
        float x = X[(size_t)row * DMODEL + idx] + Y[(size_t)row * DMODEL + idx];
        v[c] = x; s += x; ss += x * x;
    }
    #pragma unroll
    for (int off = 1; off < 32; off <<= 1) { s += __shfl_xor(s, off, 32); ss += __shfl_xor(ss, off, 32); }
    if ((tid & 31) == 0) { red[0][tid >> 5] = s; red[1][tid >> 5] = ss; }
    __syncthreads();
    float st = 0.f, sst = 0.f;
    #pragma unroll
    for (int w = 0; w < 8; w++) { st += red[0][w]; sst += red[1][w]; }
    const float mu   = st / DMODEL;
    const float var  = sst / DMODEL - mu * mu;
    const float rstd = rsqrtf(var + 1e-5f);
    #pragma unroll
    for (int c = 0; c < 4; c++) {
        int idx = tid + c * 256;
        float y = (v[c] - mu) * rstd * g[idx] + bta[idx];
        outF[(size_t)row * DMODEL + idx] = y;
        if (outB) outB[(size_t)row * DMODEL + idx] = (__bf16)y;
    }
}

// ---------------------------------------------------------------- launch
extern "C" void kernel_launch(void* const* d_in, const int* in_sizes, int n_in,
                              void* d_out, int out_size, void* d_ws, size_t ws_size,
                              hipStream_t stream)
{
    (void)in_sizes; (void)n_in; (void)out_size; (void)ws_size;
    const float* X    = (const float*)d_in[0];
    const float* Wqkv = (const float*)d_in[1];
    const float* bqkv = (const float*)d_in[2];
    const float* Wo   = (const float*)d_in[3];
    const float* bo   = (const float*)d_in[4];
    const float* W1   = (const float*)d_in[5];
    const float* b1   = (const float*)d_in[6];
    const float* W2   = (const float*)d_in[7];
    const float* b2   = (const float*)d_in[8];
    const float* ln1g = (const float*)d_in[9];
    const float* ln1b = (const float*)d_in[10];
    const float* ln2g = (const float*)d_in[11];
    const float* ln2b = (const float*)d_in[12];

    char* ws = (char*)d_ws;
    size_t off = 0;
    auto carve = [&](size_t bytes) -> char* {
        char* p = ws + off;
        off = (off + bytes + 255) & ~(size_t)255;
        return p;
    };
    __bf16* Xbf   = (__bf16*)carve((size_t)MROWS * DMODEL * 2);
    __bf16* Wqkvt = (__bf16*)carve((size_t)3 * DMODEL * DMODEL * 2);
    __bf16* Wot   = (__bf16*)carve((size_t)DMODEL * DMODEL * 2);
    __bf16* W1t   = (__bf16*)carve((size_t)DFF * DMODEL * 2);
    __bf16* W2t   = (__bf16*)carve((size_t)DMODEL * DFF * 2);
    __bf16* QKVb  = (__bf16*)carve((size_t)MROWS * 3 * DMODEL * 2);
    __bf16* Attn  = (__bf16*)carve((size_t)MROWS * DMODEL * 2);
    float*  Oproj = (float*) carve((size_t)MROWS * DMODEL * 4);
    float*  H1f   = (float*) carve((size_t)MROWS * DMODEL * 4);
    __bf16* H1b   = (__bf16*)carve((size_t)MROWS * DMODEL * 2);
    __bf16* FF1   = (__bf16*)carve((size_t)MROWS * DFF * 2);
    float*  H2f   = (float*) carve((size_t)MROWS * DMODEL * 4);

    cast_f32_bf16<<<512, 256, 0, stream>>>(X, Xbf, (size_t)MROWS * DMODEL);

    dim3 tb(32, 8);
    transpose_cast<<<dim3(3 * DMODEL / 32, DMODEL / 32), tb, 0, stream>>>(Wqkv, Wqkvt, DMODEL, 3 * DMODEL);
    transpose_cast<<<dim3(DMODEL / 32, DMODEL / 32),     tb, 0, stream>>>(Wo,   Wot,   DMODEL, DMODEL);
    transpose_cast<<<dim3(DFF / 32, DMODEL / 32),        tb, 0, stream>>>(W1,   W1t,   DMODEL, DFF);
    transpose_cast<<<dim3(DMODEL / 32, DFF / 32),        tb, 0, stream>>>(W2,   W2t,   DFF,    DMODEL);

    gemm_bf16_wmma<0><<<dim3(3 * DMODEL / 64, MROWS / 128), 256, 0, stream>>>(
        Xbf, Wqkvt, bqkv, nullptr, QKVb, MROWS, 3 * DMODEL, DMODEL);

    attention_wmma<<<dim3(SEQB * NHEADS, SEQT / 128), 256, 0, stream>>>(QKVb, Attn);

    gemm_bf16_wmma<2><<<dim3(DMODEL / 64, MROWS / 128), 256, 0, stream>>>(
        Attn, Wot, bo, Oproj, nullptr, MROWS, DMODEL, DMODEL);

    add_ln<<<MROWS, 256, 0, stream>>>(X, Oproj, ln1g, ln1b, H1f, H1b);

    gemm_bf16_wmma<1><<<dim3(DFF / 64, MROWS / 128), 256, 0, stream>>>(
        H1b, W1t, b1, nullptr, FF1, MROWS, DFF, DMODEL);

    gemm_bf16_wmma<2><<<dim3(DMODEL / 64, MROWS / 128), 256, 0, stream>>>(
        FF1, W2t, b2, H2f, nullptr, MROWS, DMODEL, DFF);

    add_ln<<<MROWS, 256, 0, stream>>>(H1f, H2f, ln2g, ln2b, (float*)d_out, nullptr);
}
